// GAT_10977936408711
// MI455X (gfx1250) — matrix operand order repaired and checked by reading, not compiled, not run on hardware
//
#include <hip/hip_runtime.h>

// GAT, 2 layers, N=4096, FIN=512, HID=256 (H=4 x D=64), C=64.
// f16 WMMA (fp32 accum) for feature GEMMs and neighbor-aggregation GEMMs.
// Two-pass fp32 softmax stats -> fully normalized P tiles (no rescale).
// All WMMA B-operands are pre-packed fragment-contiguous so every fragment
// load is two global_load_b128's.

typedef __attribute__((ext_vector_type(16))) _Float16 v16h;
typedef __attribute__((ext_vector_type(8)))  _Float16 v8h;
typedef __attribute__((ext_vector_type(8)))  float    v8f;

#define GAT_N    4096
#define GAT_FIN  512
#define GAT_HID  256
#define GAT_H    4
#define GAT_C    64
#define GAT_NEG  0.2f

// ---------------------------------------------------------------- f32 -> f16
__global__ void k_f32_to_f16(const float* __restrict__ in,
                             _Float16* __restrict__ out, int n) {
  int i = blockIdx.x * blockDim.x + threadIdx.x;
  if (i < n) out[i] = (_Float16)in[i];
}

// -------------------------------------------------- pack f32 B -> f16 B-frag
// Fragment layout: Bp[((tk*ntn + tn)*32 + lane)*16 + elem]
//   tk = k>>5, tn = n>>4, lane = (n&15) + 2*(k&8), elem = (k&7) + ((k&16)>>1)
// so a lane's whole v16h B-fragment is 32 contiguous bytes.
__global__ void k_pack_b(const float* __restrict__ B, _Float16* __restrict__ Bp,
                         int K, int Nn) {
  int idx = blockIdx.x * blockDim.x + threadIdx.x;
  if (idx >= K * Nn) return;
  int k  = idx / Nn, nc = idx % Nn;
  int tk = k >> 5,  kk = k & 31;
  int tn = nc >> 4, nn = nc & 15;
  int lane = nn + ((kk & 8) << 1);
  int elem = (kk & 7) + ((kk & 16) >> 1);
  int ntn  = Nn >> 4;
  Bp[(((size_t)(tk * ntn + tn) * 32 + lane) << 4) + elem] = (_Float16)B[idx];
}

__device__ __forceinline__ v16h load_frag(const _Float16* p) {
  v8h lo = *(const v8h*)p;
  v8h hi = *(const v8h*)(p + 8);
  return __builtin_shufflevector(lo, hi,
      0, 1, 2, 3, 4, 5, 6, 7, 8, 9, 10, 11, 12, 13, 14, 15);
}

// --------------------------------------------- C[M,Nn] = A[M,K] x B[K,Nn]
// A: row-major f16.  Bp: packed fragments.  One wave per 16x64 C tile.
__global__ void k_gemm_f16(const _Float16* __restrict__ A,
                           const _Float16* __restrict__ Bp,
                           float* __restrict__ C, int M, int K, int Nn) {
  int nt64  = Nn >> 6;
  int tile  = blockIdx.x;
  int i0    = (tile / nt64) << 4;
  int n0    = (tile % nt64) << 6;
  int lane  = threadIdx.x & 31;
  int r     = lane & 15;
  int hi    = lane >> 4;
  int abase = hi ? 8 : 0;
  int ntn   = Nn >> 4;

  const _Float16* arow = A + (size_t)(i0 + r) * K;
  v8f acc0 = {}, acc1 = {}, acc2 = {}, acc3 = {};
  for (int kt = 0; kt < K; kt += 32) {
    v8h alo = *(const v8h*)(arow + kt + abase);
    v8h ahi = *(const v8h*)(arow + kt + 16 + abase);
    v16h af = __builtin_shufflevector(alo, ahi,
        0, 1, 2, 3, 4, 5, 6, 7, 8, 9, 10, 11, 12, 13, 14, 15);
    const _Float16* bp =
        Bp + (((size_t)((kt >> 5) * ntn + (n0 >> 4)) * 32 + lane) << 4);
    v16h b0 = load_frag(bp);
    v16h b1 = load_frag(bp + 512);
    v16h b2 = load_frag(bp + 1024);
    v16h b3 = load_frag(bp + 1536);
    acc0 = __builtin_amdgcn_wmma_f32_16x16x32_f16(false, af, false, b0, (short)0, acc0, false, false);
    acc1 = __builtin_amdgcn_wmma_f32_16x16x32_f16(false, af, false, b1, (short)0, acc1, false, false);
    acc2 = __builtin_amdgcn_wmma_f32_16x16x32_f16(false, af, false, b2, (short)0, acc2, false, false);
    acc3 = __builtin_amdgcn_wmma_f32_16x16x32_f16(false, af, false, b3, (short)0, acc3, false, false);
  }
#pragma unroll
  for (int v = 0; v < 8; ++v) {
    size_t ob = (size_t)(i0 + v + hi * 8) * Nn + n0 + r;
    C[ob]      = acc0[v];
    C[ob + 16] = acc1[v];
    C[ob + 32] = acc2[v];
    C[ob + 48] = acc3[v];
  }
}

// --------------------------------------------- per-(node,head) attention dots
__global__ void k_scores(const float* __restrict__ Hmat,
                         const float* __restrict__ a_src,
                         const float* __restrict__ a_dst,
                         float* __restrict__ s_src_t,
                         float* __restrict__ s_dst_t,
                         int n_nodes, int hidtot) {
  int n = blockIdx.x, h = blockIdx.y, lane = threadIdx.x;
  const float* row = Hmat + (size_t)n * hidtot + h * 64;
  const float* as  = a_src + h * 64;
  const float* ad  = a_dst + h * 64;
  float x0 = row[lane], x1 = row[lane + 32];
  float ps = x0 * as[lane] + x1 * as[lane + 32];
  float pd = x0 * ad[lane] + x1 * ad[lane + 32];
#pragma unroll
  for (int off = 16; off >= 1; off >>= 1) {
    ps += __shfl_xor(ps, off, 32);
    pd += __shfl_xor(pd, off, 32);
  }
  if (lane == 0) {
    s_src_t[(size_t)h * n_nodes + n] = ps;
    s_dst_t[(size_t)h * n_nodes + n] = pd;
  }
}

// --------------------------------------------- pass A: softmax stats (m, z)
__device__ __forceinline__ void stat_update(int a, float sdv, float ss,
                                            float& m, float& z) {
  float e;
  if (a > 0) {
    float t = ss + sdv;
    e = (t >= 0.f) ? t : GAT_NEG * t;
  } else {
    e = -1e9f;
  }
  if (e > m) { z *= __expf(m - e); m = e; }
  z += __expf(e - m);
}

__global__ void k_softmax_stats(const int* __restrict__ A,
                                const float* __restrict__ s_src_t,
                                const float* __restrict__ s_dst_t,
                                float* __restrict__ m_t,
                                float* __restrict__ z_t, int n_nodes) {
  int i    = blockIdx.x;
  int h    = threadIdx.x >> 5;
  int lane = threadIdx.x & 31;
  size_t hN = (size_t)h * n_nodes;
  float ss = s_src_t[hN + i];
  const int*   arow = A + (size_t)i * n_nodes;
  const float* sd   = s_dst_t + hN;
  float m = -1e9f, z = 0.f;
  for (int j = lane * 4; j < n_nodes; j += 128) {
    int4   av = *(const int4*)(arow + j);
    float4 sv = *(const float4*)(sd + j);
    stat_update(av.x, sv.x, ss, m, z);
    stat_update(av.y, sv.y, ss, m, z);
    stat_update(av.z, sv.z, ss, m, z);
    stat_update(av.w, sv.w, ss, m, z);
  }
#pragma unroll
  for (int off = 16; off >= 1; off >>= 1) {
    float mo = __shfl_xor(m, off, 32);
    float zo = __shfl_xor(z, off, 32);
    float M  = fmaxf(m, mo);
    z = z * __expf(m - M) + zo * __expf(mo - M);
    m = M;
  }
  if (lane == 0) { m_t[hN + i] = m; z_t[hN + i] = z; }
}

// --------------------------------------------- pass B: fused P x V aggregation
// Block = 4 waves, all on the same (16-row i-tile, head); waves stripe the
// j-range, accumulate private 16x64 WMMA tiles, then LDS-reduce + epilogue.
__device__ __forceinline__ float prob_of(int a, float sdv, float ss, float mi,
                                         float rz) {
  float t = ss + sdv;
  t = (t >= 0.f) ? t : GAT_NEG * t;
  float p = __expf(t - mi) * rz;
  return (a > 0) ? p : 0.f;
}

__global__ void k_aggregate(const int* __restrict__ A,
                            const _Float16* __restrict__ Vp,
                            const float* __restrict__ s_src_t,
                            const float* __restrict__ s_dst_t,
                            const float* __restrict__ m_t,
                            const float* __restrict__ z_t,
                            _Float16* __restrict__ outh,
                            float* __restrict__ outf,
                            int n_nodes, int vs, int apply_elu) {
  __shared__ float red[4][16 * 64];
  int i0    = blockIdx.x << 4;
  int h     = blockIdx.y;
  int w     = threadIdx.x >> 5;     // j-stripe (wave) id
  int lane  = threadIdx.x & 31;
  int r     = lane & 15;
  int hi    = lane >> 4;
  int abase = hi ? 8 : 0;
  size_t hN = (size_t)h * n_nodes;
  int row   = i0 + r;
  float ss  = s_src_t[hN + row];
  float mi  = m_t[hN + row];
  float rz  = 1.0f / z_t[hN + row];
  const int*   arow = A + (size_t)row * n_nodes;
  const float* sd   = s_dst_t + hN;
  int ntn = vs >> 4;
  int tn0 = h * 4;

  v8f acc0 = {}, acc1 = {}, acc2 = {}, acc3 = {};

  for (int j0 = w * 32; j0 < n_nodes; j0 += 128) {
    __builtin_prefetch(arow + j0 + 512, 0, 0);
    // adjacency + dst-score streams: 16B vector loads
    int4   a0 = *(const int4*)(arow + j0 + abase);
    int4   a1 = *(const int4*)(arow + j0 + abase + 4);
    int4   a2 = *(const int4*)(arow + j0 + 16 + abase);
    int4   a3 = *(const int4*)(arow + j0 + 16 + abase + 4);
    float4 s0 = *(const float4*)(sd + j0 + abase);
    float4 s1 = *(const float4*)(sd + j0 + abase + 4);
    float4 s2 = *(const float4*)(sd + j0 + 16 + abase);
    float4 s3 = *(const float4*)(sd + j0 + 16 + abase + 4);
    v16h pf;
    pf[0]  = (_Float16)prob_of(a0.x, s0.x, ss, mi, rz);
    pf[1]  = (_Float16)prob_of(a0.y, s0.y, ss, mi, rz);
    pf[2]  = (_Float16)prob_of(a0.z, s0.z, ss, mi, rz);
    pf[3]  = (_Float16)prob_of(a0.w, s0.w, ss, mi, rz);
    pf[4]  = (_Float16)prob_of(a1.x, s1.x, ss, mi, rz);
    pf[5]  = (_Float16)prob_of(a1.y, s1.y, ss, mi, rz);
    pf[6]  = (_Float16)prob_of(a1.z, s1.z, ss, mi, rz);
    pf[7]  = (_Float16)prob_of(a1.w, s1.w, ss, mi, rz);
    pf[8]  = (_Float16)prob_of(a2.x, s2.x, ss, mi, rz);
    pf[9]  = (_Float16)prob_of(a2.y, s2.y, ss, mi, rz);
    pf[10] = (_Float16)prob_of(a2.z, s2.z, ss, mi, rz);
    pf[11] = (_Float16)prob_of(a2.w, s2.w, ss, mi, rz);
    pf[12] = (_Float16)prob_of(a3.x, s3.x, ss, mi, rz);
    pf[13] = (_Float16)prob_of(a3.y, s3.y, ss, mi, rz);
    pf[14] = (_Float16)prob_of(a3.z, s3.z, ss, mi, rz);
    pf[15] = (_Float16)prob_of(a3.w, s3.w, ss, mi, rz);
    // packed V fragments: contiguous 32B per lane per fragment
    const _Float16* vp =
        Vp + (((size_t)((j0 >> 5) * ntn + tn0) * 32 + lane) << 4);
    v16h b0 = load_frag(vp);
    v16h b1 = load_frag(vp + 512);
    v16h b2 = load_frag(vp + 1024);
    v16h b3 = load_frag(vp + 1536);
    acc0 = __builtin_amdgcn_wmma_f32_16x16x32_f16(false, pf, false, b0, (short)0, acc0, false, false);
    acc1 = __builtin_amdgcn_wmma_f32_16x16x32_f16(false, pf, false, b1, (short)0, acc1, false, false);
    acc2 = __builtin_amdgcn_wmma_f32_16x16x32_f16(false, pf, false, b2, (short)0, acc2, false, false);
    acc3 = __builtin_amdgcn_wmma_f32_16x16x32_f16(false, pf, false, b3, (short)0, acc3, false, false);
  }

  // LDS cross-wave reduction of the four j-stripe partials.
#pragma unroll
  for (int v = 0; v < 8; ++v) {
    int rr = (v + hi * 8) * 64;
    red[w][rr + r]      = acc0[v];
    red[w][rr + 16 + r] = acc1[v];
    red[w][rr + 32 + r] = acc2[v];
    red[w][rr + 48 + r] = acc3[v];
  }
  __syncthreads();
  int tid = threadIdx.x;
#pragma unroll
  for (int e = 0; e < 8; ++e) {
    int idx = tid + e * 128;
    float y = red[0][idx] + red[1][idx] + red[2][idx] + red[3][idx];
    if (apply_elu) y = (y > 0.f) ? y : __expf(y) - 1.f;
    int rr = idx >> 6, cc = idx & 63;
    size_t ob = (size_t)(i0 + rr) * vs + h * 64 + cc;
    if (outh) outh[ob] = (_Float16)y;
    if (outf) outf[ob] = y;
  }
}

// ---------------------------------------------------------------- launcher
extern "C" void kernel_launch(void* const* d_in, const int* in_sizes, int n_in,
                              void* d_out, int out_size, void* d_ws, size_t ws_size,
                              hipStream_t stream) {
  (void)in_sizes; (void)n_in; (void)out_size; (void)ws_size;
  const float* x      = (const float*)d_in[0];
  const int*   A      = (const int*)  d_in[1];
  const float* W1     = (const float*)d_in[2];
  const float* a1_src = (const float*)d_in[3];
  const float* a1_dst = (const float*)d_in[4];
  const float* W2     = (const float*)d_in[5];
  const float* a2_src = (const float*)d_in[6];
  const float* a2_dst = (const float*)d_in[7];
  float* out = (float*)d_out;

  char* p = (char*)d_ws;
  auto carve = [&](size_t bytes) -> char* {
    char* q = p;
    p += (bytes + 255) & ~(size_t)255;
    return q;
  };
  _Float16* xh  = (_Float16*)carve((size_t)GAT_N * GAT_FIN * 2);
  _Float16* w1p = (_Float16*)carve((size_t)GAT_FIN * GAT_HID * 2);  // packed
  float*    h1f = (float*)   carve((size_t)GAT_N * GAT_HID * 4);
  _Float16* v1p = (_Float16*)carve((size_t)GAT_N * GAT_HID * 2);    // packed V
  float*    s1s = (float*)   carve((size_t)GAT_H * GAT_N * 4);
  float*    s1d = (float*)   carve((size_t)GAT_H * GAT_N * 4);
  float*    m1  = (float*)   carve((size_t)GAT_H * GAT_N * 4);
  float*    z1  = (float*)   carve((size_t)GAT_H * GAT_N * 4);
  _Float16* g1h = (_Float16*)carve((size_t)GAT_N * GAT_HID * 2);    // elu out f16
  _Float16* w2p = (_Float16*)carve((size_t)GAT_HID * GAT_C * 2);    // packed
  float*    h2f = (float*)   carve((size_t)GAT_N * GAT_C * 4);
  _Float16* v2p = (_Float16*)carve((size_t)GAT_N * GAT_C * 2);      // packed V
  float*    s2s = (float*)   carve((size_t)GAT_N * 4);
  float*    s2d = (float*)   carve((size_t)GAT_N * 4);
  float*    m2  = (float*)   carve((size_t)GAT_N * 4);
  float*    z2  = (float*)   carve((size_t)GAT_N * 4);

  int n;
  // ---- layer 1
  n = GAT_N * GAT_FIN;
  k_f32_to_f16<<<(n + 255) / 256, 256, 0, stream>>>(x, xh, n);
  n = GAT_FIN * GAT_HID;
  k_pack_b<<<(n + 255) / 256, 256, 0, stream>>>(W1, w1p, GAT_FIN, GAT_HID);
  k_gemm_f16<<<(GAT_N / 16) * (GAT_HID / 64), 32, 0, stream>>>(
      xh, w1p, h1f, GAT_N, GAT_FIN, GAT_HID);
  n = GAT_N * GAT_HID;
  k_pack_b<<<(n + 255) / 256, 256, 0, stream>>>(h1f, v1p, GAT_N, GAT_HID);
  k_scores<<<dim3(GAT_N, GAT_H), 32, 0, stream>>>(
      h1f, a1_src, a1_dst, s1s, s1d, GAT_N, GAT_HID);
  k_softmax_stats<<<GAT_N, 32 * GAT_H, 0, stream>>>(A, s1s, s1d, m1, z1, GAT_N);
  k_aggregate<<<dim3(GAT_N / 16, GAT_H), 128, 0, stream>>>(
      A, v1p, s1s, s1d, m1, z1, g1h, (float*)nullptr, GAT_N, GAT_HID, 1);

  // ---- layer 2 (H=1, D=C=64)
  n = GAT_HID * GAT_C;
  k_pack_b<<<(n + 255) / 256, 256, 0, stream>>>(W2, w2p, GAT_HID, GAT_C);
  k_gemm_f16<<<(GAT_N / 16) * (GAT_C / 64), 32, 0, stream>>>(
      g1h, w2p, h2f, GAT_N, GAT_HID, GAT_C);
  n = GAT_N * GAT_C;
  k_pack_b<<<(n + 255) / 256, 256, 0, stream>>>(h2f, v2p, GAT_N, GAT_C);
  k_scores<<<dim3(GAT_N, 1), 32, 0, stream>>>(
      h2f, a2_src, a2_dst, s2s, s2d, GAT_N, GAT_C);
  k_softmax_stats<<<GAT_N, 32, 0, stream>>>(A, s2s, s2d, m2, z2, GAT_N);
  k_aggregate<<<dim3(GAT_N / 16, 1), 128, 0, stream>>>(
      A, v2p, s2s, s2d, m2, z2, (_Float16*)nullptr, out, GAT_N, GAT_C, 0);
}
